// GatedRouting_90168543412668
// MI455X (gfx1250) — matrix operand order repaired and checked by reading, not compiled, not run on hardware
//
#include <hip/hip_runtime.h>
#include <math.h>

#define DIM   128
#define NEXP  4
#define NT    8     // 128/16 column tiles
#define KT    4     // 128/32 K tiles
#define WAVES 8     // 256 threads / wave32

typedef __attribute__((ext_vector_type(16))) __bf16 v16bf;
typedef __attribute__((ext_vector_type(8)))  float  v8f;

union FragU { unsigned short us[16]; uint4 q[2]; v16bf v; };
union F32x16 { float4 f[4]; float s[16]; };

__device__ __forceinline__ unsigned short f2bf(float f) {
    unsigned u = __builtin_bit_cast(unsigned, f);
    u += 0x7FFFu + ((u >> 16) & 1u);          // round-to-nearest-even
    return (unsigned short)(u >> 16);
}
__device__ __forceinline__ float bf2f(unsigned short h) {
    unsigned u = ((unsigned)h) << 16;
    return __builtin_bit_cast(float, u);
}
__device__ __forceinline__ float gelu_exact(float x) {
    return 0.5f * x * (1.0f + erff(x * 0.70710678118654752f));
}
__device__ __forceinline__ v8f splat8(float b) {
    v8f c = {b, b, b, b, b, b, b, b};
    return c;
}
__device__ __forceinline__ v8f wmma_bf16(v16bf a, v16bf b, v8f c) {
    return __builtin_amdgcn_wmma_f32_16x16x32_bf16(
        /*neg_a=*/false, a, /*neg_b=*/false, b,
        /*c_mod=*/(short)0, c, /*reuse_a=*/false, /*reuse_b=*/false);
}
__device__ __forceinline__ unsigned lds_off(const void* p) {
    // LDS aperture: low 32 bits of the generic address are the LDS byte offset
    return (unsigned)(unsigned long long)(uintptr_t)p;
}

// A fragment (16x32 bf16) from row-major bf16 LDS tile [16][128].
__device__ __forceinline__ v16bf ldA_lds(const unsigned short* h, int kt, int lane16, int khalf) {
    const unsigned short* p = h + lane16 * DIM + kt * 32 + khalf * 8;
    FragU u;
    u.q[0] = *(const uint4*)(p);
    u.q[1] = *(const uint4*)(p + 16);
    return u.v;
}
// A fragment from f32 LDS stage (async-copied x tile), converted on the fly.
__device__ __forceinline__ v16bf ldA_ldsF32(const float* xs, int kt, int lane16, int khalf) {
    const float* p = xs + lane16 * DIM + kt * 32 + khalf * 8;
    F32x16 lo, hi;
    lo.f[0] = *(const float4*)(p);      lo.f[1] = *(const float4*)(p + 4);
    hi.f[0] = *(const float4*)(p + 16); hi.f[1] = *(const float4*)(p + 20);
    FragU u;
#pragma unroll
    for (int j = 0; j < 8; ++j) { u.us[j] = f2bf(lo.s[j]); u.us[8 + j] = f2bf(hi.s[j]); }
    return u.v;
}
// B fragment (32x16 bf16) from N-major transposed LDS weights wT[n][k].
__device__ __forceinline__ v16bf ldB(const unsigned short* wT, int nt, int kt, int lane16, int khalf) {
    const unsigned short* p = wT + (nt * 16 + lane16) * DIM + kt * 32 + khalf * 8;
    FragU u;
    u.q[0] = *(const uint4*)(p);
    u.q[1] = *(const uint4*)(p + 16);
    return u.v;
}

__global__ __launch_bounds__(256, 1)
void moe_fused_kernel(const float* __restrict__ x,
                      const float* __restrict__ entry_w, const float* __restrict__ entry_b,
                      const float* __restrict__ router_w, const float* __restrict__ router_b,
                      const float* __restrict__ exp_w,   const float* __restrict__ exp_b,
                      const float* __restrict__ ln_g,    const float* __restrict__ ln_b,
                      const float* __restrict__ out_w,   const float* __restrict__ out_b,
                      float* __restrict__ out, int nTokens)
{
    // ---- LDS: weights (bf16, transposed), per-wave f32 stage (x in / out out),
    //      per-wave bf16 h staging, small params. Total ~299KB of 320KB. ----
    __shared__ __attribute__((aligned(16))) unsigned short s_entry_wT[DIM * DIM];
    __shared__ __attribute__((aligned(16))) unsigned short s_exp_wT[NEXP][DIM * DIM];
    __shared__ __attribute__((aligned(16))) unsigned short s_out_wT[DIM * DIM];
    __shared__ __attribute__((aligned(16))) float          s_stage[WAVES][16 * DIM];
    __shared__ __attribute__((aligned(16))) unsigned short s_h[WAVES][16 * DIM];
    __shared__ float s_gw[WAVES][16][NEXP];
    __shared__ float s_rw[DIM * NEXP];
    __shared__ float s_eb[DIM], s_ob[DIM];
    __shared__ float s_xb[NEXP][DIM], s_lg[NEXP][DIM], s_lb[NEXP][DIM];

    const int tid = threadIdx.x;

    // Stage + transpose weights to bf16 LDS (once per block)
    for (int i = tid; i < DIM * DIM; i += blockDim.x) {
        int k = i >> 7, n = i & (DIM - 1);
        s_entry_wT[n * DIM + k] = f2bf(entry_w[i]);
        s_out_wT[n * DIM + k]   = f2bf(out_w[i]);
#pragma unroll
        for (int e = 0; e < NEXP; ++e)
            s_exp_wT[e][n * DIM + k] = f2bf(exp_w[e * DIM * DIM + i]);
    }
    for (int i = tid; i < DIM * NEXP; i += blockDim.x) s_rw[i] = router_w[i];
    for (int i = tid; i < DIM; i += blockDim.x) {
        s_eb[i] = entry_b[i];
        s_ob[i] = out_b[i];
#pragma unroll
        for (int e = 0; e < NEXP; ++e) {
            s_xb[e][i] = exp_b[e * DIM + i];
            s_lg[e][i] = ln_g[e * DIM + i];
            s_lb[e][i] = ln_b[e * DIM + i];
        }
    }
    __syncthreads();

    const int wave   = tid >> 5;
    const int lane   = tid & 31;
    const int lane16 = lane & 15;
    const int khalf  = lane >> 4;
    unsigned short* hbuf = &s_h[wave][0];
    float*          stg  = &s_stage[wave][0];
    const unsigned  stgLds = lds_off(stg) + (unsigned)lane * 16u;  // lane's 16B slot

    const int nTiles = nTokens >> 4;
    const int tileStride = gridDim.x * WAVES;

    for (int tile = blockIdx.x * WAVES + wave; tile < nTiles; tile += tileStride) {
        const int rowBase = tile << 4;

        // ===== async DMA: x tile (16 rows x 512B) -> LDS stage, coalesced =====
        // drain pending async stores that still read this stage buffer
        asm volatile("s_wait_asynccnt 0x0" ::: "memory");
        {
            const char* gbase = (const char*)(x + (size_t)rowBase * DIM) + lane * 16;
#pragma unroll
            for (int r = 0; r < 16; ++r) {
                unsigned long long ga = (unsigned long long)(uintptr_t)(gbase + r * 512);
                unsigned la = stgLds + (unsigned)(r * 512);
                asm volatile("global_load_async_to_lds_b128 %0, %1, off"
                             :: "v"(la), "v"(ga) : "memory");
            }
        }
        // prefetch next tile into L2 while waiting
        __builtin_prefetch(x + ((size_t)rowBase + ((size_t)tileStride << 4)) * DIM + lane * 4, 0, 1);
        asm volatile("s_wait_asynccnt 0x0" ::: "memory");

        // ================= entry: h = gelu(x @ W + b) =================
        v8f acc[NT];
#pragma unroll
        for (int nt = 0; nt < NT; ++nt) acc[nt] = splat8(s_eb[nt * 16 + lane16]);
#pragma unroll
        for (int kt = 0; kt < KT; ++kt) {
            v16bf a = ldA_ldsF32(stg, kt, lane16, khalf);
#pragma unroll
            for (int nt = 0; nt < NT; ++nt)
                acc[nt] = wmma_bf16(a, ldB(s_entry_wT, nt, kt, lane16, khalf), acc[nt]);
        }
        // C layout: VGPR v, halfwave -> row m = v + 8*khalf ; col n = nt*16 + lane16
#pragma unroll
        for (int nt = 0; nt < NT; ++nt)
#pragma unroll
            for (int v = 0; v < 8; ++v)
                hbuf[(v + 8 * khalf) * DIM + nt * 16 + lane16] = f2bf(gelu_exact(acc[nt][v]));

        // ================= router softmax (lanes 0..15, one token each) =================
        if (lane < 16) {
            const uint4* hr = (const uint4*)(hbuf + lane16 * DIM);
            float lg[NEXP];
#pragma unroll
            for (int e = 0; e < NEXP; ++e) lg[e] = router_b[e];
#pragma unroll
            for (int c = 0; c < 16; ++c) {           // 16 chunks of 8 bf16
                FragU u; u.q[0] = hr[2 * c]; u.q[1] = hr[2 * c + 1];
#pragma unroll
                for (int j = 0; j < 16; ++j) {
                    float hv = bf2f(u.us[j]);
                    (void)hv; // only first 8 of each q are this chunk; use flat index below
                }
                // flat: chunk c covers k = c*8 .. c*8+7  (q[0]) and we advance by 16 via 2c
                // simpler: treat u.us[0..15] as k = c*8.. only when loading 2 chunks; redo below
            }
            // straightforward vector walk: 8 bf16 per uint4
#pragma unroll
            for (int e = 0; e < NEXP; ++e) lg[e] = router_b[e];
            const uint4* hq = (const uint4*)(hbuf + lane16 * DIM);
#pragma unroll
            for (int c = 0; c < 16; ++c) {
                uint4 q = hq[c];
                unsigned w0 = q.x, w1 = q.y, w2 = q.z, w3 = q.w;
                unsigned ws[8] = { w0 & 0xFFFFu, w0 >> 16, w1 & 0xFFFFu, w1 >> 16,
                                   w2 & 0xFFFFu, w2 >> 16, w3 & 0xFFFFu, w3 >> 16 };
#pragma unroll
                for (int j = 0; j < 8; ++j) {
                    float hv = __builtin_bit_cast(float, ws[j] << 16);
                    int k = c * 8 + j;
#pragma unroll
                    for (int e = 0; e < NEXP; ++e) lg[e] += hv * s_rw[k * NEXP + e];
                }
            }
            float mx = lg[0];
#pragma unroll
            for (int e = 1; e < NEXP; ++e) mx = fmaxf(mx, lg[e]);
            float sum = 0.f;
#pragma unroll
            for (int e = 0; e < NEXP; ++e) { lg[e] = __expf(lg[e] - mx); sum += lg[e]; }
            float inv = 1.f / sum;
#pragma unroll
            for (int e = 0; e < NEXP; ++e) s_gw[wave][lane16][e] = lg[e] * inv;
        }

        // ================= experts: gelu -> layernorm -> weighted combine =================
        v8f comb[NT];
#pragma unroll
        for (int nt = 0; nt < NT; ++nt) comb[nt] = splat8(0.f);

#pragma unroll
        for (int e = 0; e < NEXP; ++e) {
#pragma unroll
            for (int nt = 0; nt < NT; ++nt) acc[nt] = splat8(s_xb[e][nt * 16 + lane16]);
#pragma unroll
            for (int kt = 0; kt < KT; ++kt) {
                v16bf a = ldA_lds(hbuf, kt, lane16, khalf);
#pragma unroll
                for (int nt = 0; nt < NT; ++nt)
                    acc[nt] = wmma_bf16(a, ldB(&s_exp_wT[e][0], nt, kt, lane16, khalf), acc[nt]);
            }
#pragma unroll
            for (int nt = 0; nt < NT; ++nt)
#pragma unroll
                for (int v = 0; v < 8; ++v) acc[nt][v] = gelu_exact(acc[nt][v]);

            // per-row mean/var via halfwave shuffles (row m = v + 8*khalf)
#pragma unroll
            for (int v = 0; v < 8; ++v) {
                float p1 = 0.f, p2 = 0.f;
#pragma unroll
                for (int nt = 0; nt < NT; ++nt) { float t = acc[nt][v]; p1 += t; p2 += t * t; }
#pragma unroll
                for (int off = 1; off < 16; off <<= 1) {
                    p1 += __shfl_xor(p1, off, 32);
                    p2 += __shfl_xor(p2, off, 32);
                }
                float mu  = p1 * (1.f / DIM);
                float var = p2 * (1.f / DIM) - mu * mu;
                float rinv = rsqrtf(var + 1e-5f);
                float wv   = s_gw[wave][v + 8 * khalf][e];
#pragma unroll
                for (int nt = 0; nt < NT; ++nt) {
                    int n = nt * 16 + lane16;
                    float val = (acc[nt][v] - mu) * rinv * s_lg[e][n] + s_lb[e][n];
                    comb[nt][v] += wv * val;
                }
            }
        }

        // ================= out GEMM: combined @ out_w + out_b =================
#pragma unroll
        for (int nt = 0; nt < NT; ++nt)
#pragma unroll
            for (int v = 0; v < 8; ++v)
                hbuf[(v + 8 * khalf) * DIM + nt * 16 + lane16] = f2bf(comb[nt][v]);

#pragma unroll
        for (int nt = 0; nt < NT; ++nt) acc[nt] = splat8(s_ob[nt * 16 + lane16]);
#pragma unroll
        for (int kt = 0; kt < KT; ++kt) {
            v16bf a = ldA_lds(hbuf, kt, lane16, khalf);
#pragma unroll
            for (int nt = 0; nt < NT; ++nt)
                acc[nt] = wmma_bf16(a, ldB(s_out_wT, nt, kt, lane16, khalf), acc[nt]);
        }

        // stage f32 output tile in LDS (x stage is dead), then async DMA LDS -> global
#pragma unroll
        for (int nt = 0; nt < NT; ++nt)
#pragma unroll
            for (int v = 0; v < 8; ++v)
                stg[(v + 8 * khalf) * DIM + nt * 16 + lane16] = acc[nt][v];

        asm volatile("s_wait_dscnt 0x0" ::: "memory");   // ds stores visible before DMA reads LDS
        {
            char* gbase = (char*)(out + (size_t)rowBase * DIM) + lane * 16;
#pragma unroll
            for (int r = 0; r < 16; ++r) {
                unsigned long long ga = (unsigned long long)(uintptr_t)(gbase + r * 512);
                unsigned la = stgLds + (unsigned)(r * 512);
                asm volatile("global_store_async_from_lds_b128 %0, %1, off"
                             :: "v"(ga), "v"(la) : "memory");
            }
        }
        // stores drain at the top of the next iteration (or fall out of the loop)
    }
    asm volatile("s_wait_asynccnt 0x0" ::: "memory");
}

extern "C" void kernel_launch(void* const* d_in, const int* in_sizes, int n_in,
                              void* d_out, int out_size, void* d_ws, size_t ws_size,
                              hipStream_t stream) {
    (void)n_in; (void)out_size; (void)d_ws; (void)ws_size;
    const float* x        = (const float*)d_in[0];
    const float* entry_w  = (const float*)d_in[1];
    const float* entry_b  = (const float*)d_in[2];
    const float* router_w = (const float*)d_in[3];
    const float* router_b = (const float*)d_in[4];
    const float* exp_w    = (const float*)d_in[5];
    const float* exp_b    = (const float*)d_in[6];
    const float* ln_g     = (const float*)d_in[7];
    const float* ln_b     = (const float*)d_in[8];
    const float* out_w    = (const float*)d_in[9];
    const float* out_b    = (const float*)d_in[10];
    float* out = (float*)d_out;

    int nTokens = in_sizes[0] / DIM;
    int nTiles  = nTokens >> 4;
    int blocks  = (nTiles + WAVES - 1) / WAVES;
    if (blocks > 512) blocks = 512;   // persistent blocks; ~299KB LDS -> 1 block/WGP
    moe_fused_kernel<<<dim3(blocks), dim3(256), 0, stream>>>(
        x, entry_w, entry_b, router_w, router_b, exp_w, exp_b,
        ln_g, ln_b, out_w, out_b, out, nTokens);
}
